// TPLinker_49486613185319
// MI455X (gfx1250) — matrix operand order repaired and checked by reading, not compiled, not run on hardware
//
#include <hip/hip_runtime.h>
#include <math.h>

typedef __attribute__((ext_vector_type(16))) _Float16 v16h;
typedef __attribute__((ext_vector_type(8)))  _Float16 v8h;
typedef __attribute__((ext_vector_type(8)))  float    v8f;

#define Hd    768
#define Sd    256
#define Bd    4
#define Rd    24
#define Pd    32896          /* S*(S+1)/2 */
#define M1    (Bd*Sd)        /* 1024 */
#define N1    (2*Hd)         /* 1536 */
#define NOUT  146            /* 2 + 72 + 72 */
#define NP    160            /* padded to 10 n-tiles */
#define NTL   (NP/16)        /* 10 */
#define KT    (Hd/32)        /* 24 k-steps */

union U16h { v16h v; v8h h[2]; };

// ---------------------------------------------------------------------------
// Pack f32 inputs into f16 operand matrices.
//  hid_h : [M1][H]   = hidden
//  w1_h  : [N1][H]   row o<768 -> fc_w[o][h]; o>=768 -> fc_w[o-768][H+h]
//  wout_h: [NP][H]   rows 0-1 h2t_w, 2-73 h2h_w, 74-145 t2t_w, rest zero
// ---------------------------------------------------------------------------
__global__ void convert_kernel(const float* __restrict__ hidden,
                               const float* __restrict__ fc_w,
                               const float* __restrict__ h2t_w,
                               const float* __restrict__ h2h_w,
                               const float* __restrict__ t2t_w,
                               _Float16* __restrict__ hid_h,
                               _Float16* __restrict__ w1_h,
                               _Float16* __restrict__ wout_h) {
  int idx = blockIdx.x * blockDim.x + threadIdx.x;
  const int n_hid = M1 * Hd;
  const int n_w1  = N1 * Hd;
  const int n_wo  = NP * Hd;
  if (idx < n_hid) {
    hid_h[idx] = (_Float16)hidden[idx];
  } else if (idx < n_hid + n_w1) {
    int t = idx - n_hid;
    int o = t / Hd, h = t - o * Hd;
    float v = (o < Hd) ? fc_w[o * N1 + h] : fc_w[(o - Hd) * N1 + Hd + h];
    w1_h[t] = (_Float16)v;
  } else if (idx < n_hid + n_w1 + n_wo) {
    int t = idx - n_hid - n_w1;
    int n = t / Hd, h = t - n * Hd;
    float v = 0.0f;
    if (n < 2)         v = h2t_w[n * Hd + h];
    else if (n < 74)   v = h2h_w[(n - 2) * Hd + h];
    else if (n < NOUT) v = t2t_w[(n - 74) * Hd + h];
    wout_h[t] = (_Float16)v;
  }
}

// ---------------------------------------------------------------------------
// triu_indices(S): row-major upper triangle, offset(i)=i*S-i*(i-1)/2
// ---------------------------------------------------------------------------
__global__ void build_idx_kernel(int* __restrict__ idx_i, int* __restrict__ idx_j) {
  int p = blockIdx.x * blockDim.x + threadIdx.x;
  if (p >= Pd) return;
  float disc = (2.0f * Sd + 1.0f) * (2.0f * Sd + 1.0f) - 8.0f * (float)p;
  int i = (int)((2.0f * Sd + 1.0f - sqrtf(disc)) * 0.5f);
  if (i < 0) i = 0;
  while (i > 0 && i * Sd - i * (i - 1) / 2 > p) --i;
  while ((i + 1) * Sd - (i + 1) * i / 2 <= p) ++i;
  int off = i * Sd - i * (i - 1) / 2;
  idx_i[p] = i;
  idx_j[p] = i + (p - off);
}

// ---------------------------------------------------------------------------
// WMMA operand loaders (16x16x32 f16, wave32 layouts per CDNA5 ISA)
// ---------------------------------------------------------------------------
__device__ __forceinline__ v16h load_a_rm(const _Float16* __restrict__ base,
                                          int ldk, int lane, int k0) {
  int m  = lane & 15;
  int sh = (lane >> 4) * 8;
  const _Float16* row = base + m * ldk;
  U16h u;
  u.h[0] = *(const v8h*)(row + k0 + sh);
  u.h[1] = *(const v8h*)(row + k0 + 16 + sh);
  return u.v;
}

__device__ __forceinline__ v16h load_b_rm(const _Float16* __restrict__ w,
                                          int ldk, int lane, int n0, int k0) {
  int n  = n0 + (lane & 15);
  int sh = (lane >> 4) * 8;
  const _Float16* src = w + n * ldk + k0 + sh;
  U16h u;
  u.h[0] = *(const v8h*)(src);
  u.h[1] = *(const v8h*)(src + 16);
  return u.v;
}

// Build the f16 A operand in-register: lane holds row m = lane%16,
// halves 0-7 = K k0+sh.., halves 8-15 = K k0+16+sh.. (sh = (lane>=16)*8).
__device__ __forceinline__ v16h make_a(const float* __restrict__ lrow,
                                       const float* __restrict__ rrow,
                                       const float* __restrict__ bias,
                                       int k0, int sh) {
  v8f L0 = *(const v8f*)(lrow + k0 + sh);
  v8f R0 = *(const v8f*)(rrow + k0 + sh);
  v8f B0 = *(const v8f*)(bias + k0 + sh);
  v8f L1 = *(const v8f*)(lrow + k0 + 16 + sh);
  v8f R1 = *(const v8f*)(rrow + k0 + 16 + sh);
  v8f B1 = *(const v8f*)(bias + k0 + 16 + sh);
  v16h a;
#pragma unroll
  for (int e = 0; e < 8; ++e) {
    a[e]     = (_Float16)tanhf(L0[e] + R0[e] + B0[e]);
    a[e + 8] = (_Float16)tanhf(L1[e] + R1[e] + B1[e]);
  }
  return a;
}

// ---------------------------------------------------------------------------
// Phase 1: Y[m][n] = sum_h hid[m][h] * W1[n][h]   (M1 x N1, K=H)
// ---------------------------------------------------------------------------
__global__ __launch_bounds__(32)
void gemm1_kernel(const _Float16* __restrict__ hid_h,
                  const _Float16* __restrict__ w1_h,
                  float* __restrict__ Y) {
  int tile = blockIdx.x;
  int mt = tile / (N1 / 16);
  int nt = tile - mt * (N1 / 16);
  int lane = threadIdx.x;
  const _Float16* abase = hid_h + (mt * 16) * Hd;
  v8f acc = {};
  for (int kk = 0; kk < KT; ++kk) {
    v16h a = load_a_rm(abase, Hd, lane, kk * 32);
    v16h b = load_b_rm(w1_h, Hd, lane, nt * 16, kk * 32);
    acc = __builtin_amdgcn_wmma_f32_16x16x32_f16(false, a, false, b,
                                                 (short)0, acc, false, false);
  }
  int n = nt * 16 + (lane & 15);
  int m0 = mt * 16 + ((lane >> 4) ? 8 : 0);
#pragma unroll
  for (int r = 0; r < 8; ++r)
    Y[(m0 + r) * N1 + n] = acc[r];
}

// ---------------------------------------------------------------------------
// Phase 2: one wave per 32 pairs (two 16-row M-tiles sharing B registers).
// A operands built in-register (tanh fused, co-executes with WMMA);
// B operand loaded once per (kk, n-tile) and reused for both M-tiles.
// ---------------------------------------------------------------------------
__global__ __launch_bounds__(32)
void pair_kernel(const float* __restrict__ Y,
                 const float* __restrict__ fc_b,
                 const _Float16* __restrict__ wout_h,
                 const int* __restrict__ idx_i,
                 const int* __restrict__ idx_j,
                 const float* __restrict__ h2t_b,
                 const float* __restrict__ h2h_b,
                 const float* __restrict__ t2t_b,
                 float* __restrict__ out) {
  __shared__ alignas(16) float stage[32 * NP];   // 20 KB

  int tile = blockIdx.x;                 // 0 .. Bd*(Pd/32)-1
  int b  = tile / (Pd / 32);
  int pt = tile - b * (Pd / 32);
  int p0 = pt * 32;
  int lane = threadIdx.x;

  int mrow = lane & 15;
  int sh   = (lane >> 4) * 8;

  // per-lane row pointers for the two M-tiles
  int pa = p0 + mrow;
  int pb = p0 + 16 + mrow;
  int i0 = idx_i[pa], j0 = idx_j[pa];
  int i1 = idx_i[pb], j1 = idx_j[pb];
  const float* lrow0 = Y + (size_t)(b * Sd + i0) * N1;        // left
  const float* rrow0 = Y + (size_t)(b * Sd + j0) * N1 + Hd;   // right
  const float* lrow1 = Y + (size_t)(b * Sd + i1) * N1;
  const float* rrow1 = Y + (size_t)(b * Sd + j1) * N1 + Hd;

  v8f zero = {};
  v8f acc0[NTL], acc1[NTL];
#pragma unroll
  for (int t = 0; t < NTL; ++t) { acc0[t] = zero; acc1[t] = zero; }

  for (int kk = 0; kk < KT; ++kk) {
    int k0 = kk * 32;
    v16h a0 = make_a(lrow0, rrow0, fc_b, k0, sh);
    v16h a1 = make_a(lrow1, rrow1, fc_b, k0, sh);
#pragma unroll
    for (int t = 0; t < NTL; ++t) {
      v16h bm = load_b_rm(wout_h, Hd, lane, t * 16, k0);   // L2-resident
      acc0[t] = __builtin_amdgcn_wmma_f32_16x16x32_f16(false, a0, false, bm,
                                                       (short)0, acc0[t],
                                                       false, false);
      acc1[t] = __builtin_amdgcn_wmma_f32_16x16x32_f16(false, a1, false, bm,
                                                       (short)0, acc1[t],
                                                       false, false);
    }
  }

  // ---- stage C tiles to LDS: stage[pair-local row][n]
  int nlo = lane & 15;
  int rbase = (lane >> 4) * 8;
#pragma unroll
  for (int t = 0; t < NTL; ++t)
#pragma unroll
    for (int r = 0; r < 8; ++r) {
      stage[(rbase + r) * NP + t * 16 + nlo]      = acc0[t][r];
      stage[(16 + rbase + r) * NP + t * 16 + nlo] = acc1[t][r];
    }
  __syncthreads();

  // ---- epilogue over 32 pairs
  const size_t OFF1 = (size_t)Bd * Pd * 2;          // start of h2h
  const size_t SZ23 = (size_t)Bd * Rd * Pd * 3;     // size of h2h (== t2t)

  {                                                 // h2t rows, one per lane
    int p = p0 + lane;
    size_t o = ((size_t)b * Pd + p) * 2;
    out[o + 0] = stage[lane * NP + 0] + h2t_b[0];
    out[o + 1] = stage[lane * NP + 1] + h2t_b[1];
  }

  // 32 pairs x (2 matrices x 24 relations) 3-way softmaxes
  for (int wk = lane; wk < 32 * 48; wk += 32) {
    int pl  = wk / 48;
    int q   = wk - pl * 48;
    int mat = q / 24;           // 0 = h2h, 1 = t2t
    int r   = q - mat * 24;
    int col = 2 + mat * 72 + r * 3;
    const float* bias = mat ? t2t_b : h2h_b;
    float x0 = stage[pl * NP + col + 0] + bias[r * 3 + 0];
    float x1 = stage[pl * NP + col + 1] + bias[r * 3 + 1];
    float x2 = stage[pl * NP + col + 2] + bias[r * 3 + 2];
    float mx = fmaxf(x0, fmaxf(x1, x2));
    float e0 = __expf(x0 - mx);
    float e1 = __expf(x1 - mx);
    float e2 = __expf(x2 - mx);
    float inv = 1.0f / (e0 + e1 + e2);
    size_t base = OFF1 + (size_t)mat * SZ23 +
                  (((size_t)b * Rd + r) * Pd + (size_t)(p0 + pl)) * 3;
    out[base + 0] = e0 * inv;
    out[base + 1] = e1 * inv;
    out[base + 2] = e2 * inv;
  }
}

// ---------------------------------------------------------------------------
extern "C" void kernel_launch(void* const* d_in, const int* in_sizes, int n_in,
                              void* d_out, int out_size, void* d_ws, size_t ws_size,
                              hipStream_t stream) {
  (void)in_sizes; (void)n_in; (void)out_size; (void)ws_size;
  const float* hidden = (const float*)d_in[0];
  const float* fc_w   = (const float*)d_in[1];
  const float* fc_b   = (const float*)d_in[2];
  const float* h2t_w  = (const float*)d_in[3];
  const float* h2t_b  = (const float*)d_in[4];
  const float* h2h_w  = (const float*)d_in[5];
  const float* h2h_b  = (const float*)d_in[6];
  const float* t2t_w  = (const float*)d_in[7];
  const float* t2t_b  = (const float*)d_in[8];
  float* out = (float*)d_out;

  // workspace carve-up (all 16B-aligned offsets)
  char* ws = (char*)d_ws;
  _Float16* hid_h  = (_Float16*)ws;  ws += (size_t)M1 * Hd * 2;   // 1.50 MB
  _Float16* w1_h   = (_Float16*)ws;  ws += (size_t)N1 * Hd * 2;   // 2.25 MB
  _Float16* wout_h = (_Float16*)ws;  ws += (size_t)NP * Hd * 2;   // 240 KB
  float*    Y      = (float*)ws;     ws += (size_t)M1 * N1 * 4;   // 6.00 MB
  int*      idx_i  = (int*)ws;       ws += (size_t)Pd * 4;
  int*      idx_j  = (int*)ws;       ws += (size_t)Pd * 4;

  const int total_conv = M1 * Hd + N1 * Hd + NP * Hd;  // 2,088,960
  convert_kernel<<<(total_conv + 255) / 256, 256, 0, stream>>>(
      hidden, fc_w, h2t_w, h2h_w, t2t_w, hid_h, w1_h, wout_h);

  build_idx_kernel<<<(Pd + 255) / 256, 256, 0, stream>>>(idx_i, idx_j);

  gemm1_kernel<<<(M1 / 16) * (N1 / 16), 32, 0, stream>>>(hid_h, w1_h, Y);

  pair_kernel<<<Bd * (Pd / 32), 32, 0, stream>>>(
      Y, fc_b, wout_h, idx_i, idx_j, h2t_b, h2h_b, t2t_b, out);
}
